// LinearClassHead_46265387712929
// MI455X (gfx1250) — compile-verified, matrix operand order checked by reading, not run on hardware
//
#include <hip/hip_runtime.h>
#include <hip/hip_bf16.h>

typedef __attribute__((ext_vector_type(16))) __bf16 v16bf;
typedef __attribute__((ext_vector_type(8)))  float  v8f;
typedef __attribute__((ext_vector_type(4)))  int    v4i;

#define HIDDEN 1024
#define HEADS  2560
#define BATCH  8192

#if __has_builtin(__builtin_amdgcn_global_load_async_to_lds_b128)
#define USE_ASYNC_LDS 1
#else
#define USE_ASYNC_LDS 0
#endif

// Cast chains: change AS on a void*, then retype within that AS.
#define GLOBAL_V4I(p) ((__attribute__((address_space(1))) v4i*)                       \
                       (__attribute__((address_space(1))) void*)(void*)(p))
#define LDS_V4I(p)    ((__attribute__((address_space(3))) v4i*)                       \
                       (__attribute__((address_space(3))) void*)(void*)(p))

__device__ __forceinline__ void wait_async_all() {
#if __has_builtin(__builtin_amdgcn_s_wait_asynccnt)
  __builtin_amdgcn_s_wait_asynccnt(0);
#else
  asm volatile("s_wait_asynccnt 0x0" ::: "memory");
#endif
}

// ---------- float32 -> bf16 (round-to-nearest-even) ----------
__device__ __forceinline__ unsigned short f2bf(float f) {
  unsigned u = __float_as_uint(f);
  u += 0x7FFFu + ((u >> 16) & 1u);
  return (unsigned short)(u >> 16);
}

// Convert x [BATCH, HIDDEN] f32 -> bf16, 8 elements per thread.
__global__ __launch_bounds__(256) void cvt_x_bf16(const float* __restrict__ x,
                                                  unsigned short* __restrict__ xb) {
  int i = blockIdx.x * blockDim.x + threadIdx.x;   // one uint4 (8 bf16) per thread
  const float4* s = (const float4*)x;
  float4 a = s[2 * i + 0];
  float4 b = s[2 * i + 1];
  uint4 o;
  o.x = (unsigned)f2bf(a.x) | ((unsigned)f2bf(a.y) << 16);
  o.y = (unsigned)f2bf(a.z) | ((unsigned)f2bf(a.w) << 16);
  o.z = (unsigned)f2bf(b.x) | ((unsigned)f2bf(b.y) << 16);
  o.w = (unsigned)f2bf(b.z) | ((unsigned)f2bf(b.w) << 16);
  ((uint4*)xb)[i] = o;
}

// Transpose + convert W [HIDDEN, HEADS] f32 -> Wt [HEADS, HIDDEN] bf16 (K-contiguous).
__global__ __launch_bounds__(256) void cvt_w_t_bf16(const float* __restrict__ W,
                                                    unsigned short* __restrict__ Wt) {
  __shared__ float tile[32][33];
  const int n0 = blockIdx.x * 32;   // HEADS dim
  const int k0 = blockIdx.y * 32;   // HIDDEN dim
  const int tx = threadIdx.x, ty = threadIdx.y;
#pragma unroll
  for (int j = 0; j < 4; ++j)
    tile[ty + j * 8][tx] = W[(size_t)(k0 + ty + j * 8) * HEADS + (n0 + tx)];
  __syncthreads();
#pragma unroll
  for (int j = 0; j < 4; ++j)
    Wt[(size_t)(n0 + ty + j * 8) * HIDDEN + (k0 + tx)] = f2bf(tile[tx][ty + j * 8]);
}

// ---------- main GEMM: out = sigmoid(Xb @ Wt^T + bias) ----------
// Block tile 128(M) x 128(N), K-step 32, double-buffered LDS, 8 waves of 32.
// Staging uses GLOBAL_LOAD_ASYNC_TO_LDS_B128 (ASYNCcnt) when available.
__global__ __launch_bounds__(256) void heads_gemm(const unsigned short* __restrict__ Xb,
                                                  const unsigned short* __restrict__ Wt,
                                                  const float* __restrict__ bias,
                                                  float* __restrict__ out) {
  constexpr int BK = 32;
  constexpr int LDSW = 40;  // row stride in shorts (80B) -> conflict-free b128 frag loads
  __shared__ unsigned short As[2][128 * LDSW];
  __shared__ unsigned short Bs[2][128 * LDSW];

  const int t     = threadIdx.x;
  const int lane  = t & 31;
  const int wave  = t >> 5;     // 0..7
  const int half  = lane >> 4;  // 0/1 -> K-base 0 or 8 per WMMA 16-bit A/B layout
  const int r     = lane & 15;  // row within 16x16 tile
  const int waveM = wave & 3;   // 4 waves along M, 32 rows each
  const int waveN = wave >> 2;  // 2 waves along N, 64 cols each
  const int mBase = blockIdx.y * 128;
  const int nBase = blockIdx.x * 128;

  // staging: threads 0..127 copy one A row, 128..255 one B row (64B = 4 x b128)
  const int  srow = t & 127;
  const bool isB  = t >= 128;
  const unsigned short* gsrc = isB ? (Wt + (size_t)(nBase + srow) * HIDDEN)
                                   : (Xb + (size_t)(mBase + srow) * HIDDEN);

  auto stage = [&](int buf, int k0) {
    const unsigned short* g = gsrc + k0;
    unsigned short* d = (isB ? &Bs[buf][srow * LDSW] : &As[buf][srow * LDSW]);
#if USE_ASYNC_LDS
#pragma unroll
    for (int j = 0; j < 4; ++j)
      __builtin_amdgcn_global_load_async_to_lds_b128(GLOBAL_V4I(g + j * 8),
                                                     LDS_V4I(d + j * 8),
                                                     /*offset=*/0, /*cpol=*/0);
#else
    const uint4* s4 = (const uint4*)g;
    uint4* d4 = (uint4*)d;
#pragma unroll
    for (int j = 0; j < 4; ++j) d4[j] = s4[j];
#endif
  };

  union Frag { uint4 u[2]; v16bf v; };
  auto loadFrag = [&](const unsigned short* base, int row) -> v16bf {
    Frag f;
    const unsigned short* p = base + row * LDSW + half * 8;  // K = kb..kb+7
    f.u[0] = *(const uint4*)(p);
    f.u[1] = *(const uint4*)(p + 16);                        // K = kb+16..kb+23
    return f.v;
  };

  v8f acc[2][4];
#pragma unroll
  for (int mt = 0; mt < 2; ++mt)
#pragma unroll
    for (int nt = 0; nt < 4; ++nt) acc[mt][nt] = v8f{};

  stage(0, 0);
#if USE_ASYNC_LDS
  wait_async_all();
#endif
  __syncthreads();

  constexpr int KSTEPS = HIDDEN / BK;  // 32
  for (int ks = 0; ks < KSTEPS; ++ks) {
    const int cur = ks & 1;
    if (ks + 1 < KSTEPS) stage(cur ^ 1, (ks + 1) * BK);  // in flight during WMMAs

    v16bf afrag[2], bfrag[4];
#pragma unroll
    for (int mt = 0; mt < 2; ++mt)
      afrag[mt] = loadFrag(&As[cur][0], waveM * 32 + mt * 16 + r);
#pragma unroll
    for (int nt = 0; nt < 4; ++nt)
      bfrag[nt] = loadFrag(&Bs[cur][0], waveN * 64 + nt * 16 + r);

#pragma unroll
    for (int mt = 0; mt < 2; ++mt)
#pragma unroll
      for (int nt = 0; nt < 4; ++nt)
        acc[mt][nt] = __builtin_amdgcn_wmma_f32_16x16x32_bf16(
            false, afrag[mt], false, bfrag[nt], (short)0, acc[mt][nt], false, false);

#if USE_ASYNC_LDS
    wait_async_all();   // next buffer fully landed in LDS
#endif
    __syncthreads();
  }

  // epilogue: bias + sigmoid, C layout: N = lane%16, M = vgpr + 8*half
#pragma unroll
  for (int nt = 0; nt < 4; ++nt) {
    const int n = nBase + waveN * 64 + nt * 16 + r;
    const float bv = bias[n];
#pragma unroll
    for (int mt = 0; mt < 2; ++mt) {
      const int m0 = mBase + waveM * 32 + mt * 16 + half * 8;
#pragma unroll
      for (int v = 0; v < 8; ++v) {
        float z = acc[mt][nt][v] + bv;
        float e = __expf(-z);
        out[(size_t)(m0 + v) * HEADS + n] = __builtin_amdgcn_rcpf(1.0f + e);
      }
    }
  }
}

extern "C" void kernel_launch(void* const* d_in, const int* in_sizes, int n_in,
                              void* d_out, int out_size, void* d_ws, size_t ws_size,
                              hipStream_t stream) {
  (void)in_sizes; (void)n_in; (void)out_size; (void)ws_size;
  const float* x = (const float*)d_in[0];
  const float* W = (const float*)d_in[1];
  const float* b = (const float*)d_in[2];
  float* out = (float*)d_out;

  unsigned short* Xb = (unsigned short*)d_ws;                 // [BATCH][HIDDEN] bf16
  unsigned short* Wt = Xb + (size_t)BATCH * HIDDEN;           // [HEADS][HIDDEN] bf16

  cvt_x_bf16<<<(BATCH * HIDDEN / 8) / 256, 256, 0, stream>>>(x, Xb);
  cvt_w_t_bf16<<<dim3(HEADS / 32, HIDDEN / 32), dim3(32, 8), 0, stream>>>(W, Wt);
  heads_gemm<<<dim3(HEADS / 128, BATCH / 128), 256, 0, stream>>>(Xb, Wt, b, out);
}
